// MoE_1554778161721
// MI455X (gfx1250) — matrix operand order repaired and checked by reading, not compiled, not run on hardware
//
#include <hip/hip_runtime.h>
#include <stdint.h>

// ---------------------------------------------------------------------------
// MoE (T=8192, D=1024, H=2816, E=8, K=2) for gfx1250.
// Routed top-2 (2/8 of reference FLOPs), bf16 WMMA GEMMs with f32 accum.
// Weights converted+transposed to bf16 once per launch (~138MB -> L2-resident).
// Async global->LDS double-buffered staging (ASYNCcnt) when available;
// spill-free synchronous fallback otherwise.
// ---------------------------------------------------------------------------

typedef __attribute__((ext_vector_type(16))) __bf16 v16bf;
typedef __attribute__((ext_vector_type(8)))  float  v8f;
typedef __attribute__((ext_vector_type(4)))  int    v4i;

constexpr int Tn = 8192;   // B*S tokens
constexpr int Dd = 1024;
constexpr int Hh = 2816;
constexpr int Ee = 8;
constexpr int Kk = 2;

// workspace offsets (bytes)
constexpr size_t OFF_XB   = 0;                                   // T*D bf16
constexpr size_t OFF_W1T  = OFF_XB   + (size_t)Tn * Dd * 2;      // E*H*D bf16 (transposed)
constexpr size_t OFF_W3T  = OFF_W1T  + (size_t)Ee * Hh * Dd * 2;
constexpr size_t OFF_W2T  = OFF_W3T  + (size_t)Ee * Hh * Dd * 2; // E*D*H bf16 (transposed)
constexpr size_t OFF_HBUF = OFF_W2T  + (size_t)Ee * Dd * Hh * 2; // T*K*H bf16
constexpr size_t OFF_PART = OFF_HBUF + (size_t)Tn * Kk * Hh * 2; // T*K*D f32
constexpr size_t OFF_CNT  = OFF_PART + (size_t)Tn * Kk * Dd * 4; // E ints (padded)
constexpr size_t OFF_LIST = OFF_CNT  + 256;                      // E*T ints (slot ids)
constexpr size_t OFF_WSL  = OFF_LIST + (size_t)Ee * Tn * 4;      // T*K f32 routing weights

// ---- async global->LDS support (gfx1250) ----------------------------------
#if defined(__AMDGCN__) && \
    __has_builtin(__builtin_amdgcn_global_load_async_to_lds_b128) && \
    __has_builtin(__builtin_amdgcn_s_wait_asynccnt)
#define HAVE_ASYNC 1
#else
#define HAVE_ASYNC 0
#endif

constexpr int NBUF = HAVE_ASYNC ? 2 : 1;

#if HAVE_ASYNC
typedef __attribute__((address_space(1))) v4i* gas_v4i_p;
typedef __attribute__((address_space(3))) v4i* las_v4i_p;
__device__ __forceinline__ void async_cp16(void* lds, const void* g) {
  // param 0 per clang diagnostic: int4 addrspace(1)* (non-const); param 1: LDS ptr
  __builtin_amdgcn_global_load_async_to_lds_b128(
      (gas_v4i_p)(v4i*)const_cast<void*>(g),
      (las_v4i_p)(v4i*)lds, 0, 0);
}
__device__ __forceinline__ void async_wait0() {
  __builtin_amdgcn_s_wait_asynccnt(0);
}
#endif

__device__ __forceinline__ uint16_t f2bf(float f) {
  uint32_t u = __builtin_bit_cast(uint32_t, f);
  u += 0x7FFFu + ((u >> 16) & 1u);           // round-to-nearest-even
  return (uint16_t)(u >> 16);
}

// ---------------------------------------------------------------------------
__global__ void zero_counts_kernel(int* cnt) {
  if (threadIdx.x < Ee) cnt[threadIdx.x] = 0;
}

// src: [E][R][C] f32 row-major  ->  dst: [E][C][R] bf16 (transposed)
__global__ void transpose_bf16_kernel(const float* __restrict__ src,
                                      uint16_t* __restrict__ dst,
                                      int R, int C) {
  __shared__ float tile[32][33];
  const int e  = blockIdx.z;
  const float*    s = src + (size_t)e * R * C;
  uint16_t*       d = dst + (size_t)e * R * C;
  const int tx = threadIdx.x, ty = threadIdx.y;
  const int c0 = blockIdx.x * 32, r0 = blockIdx.y * 32;
#pragma unroll
  for (int j = 0; j < 32; j += 8)
    tile[ty + j][tx] = s[(size_t)(r0 + ty + j) * C + (c0 + tx)];
  __syncthreads();
#pragma unroll
  for (int j = 0; j < 32; j += 8)
    d[(size_t)(c0 + ty + j) * R + (r0 + tx)] = f2bf(tile[tx][ty + j]);
}

// ---------------------------------------------------------------------------
// Router: one wave32 per token. fp32 gate scores, top-2 + softmax, per-expert
// slot lists (atomic append), fused x -> bf16 conversion.
__global__ __launch_bounds__(256) void router_kernel(
    const float* __restrict__ x, const float* __restrict__ gw,
    uint16_t* __restrict__ xb, int* __restrict__ cnt,
    int* __restrict__ list, float* __restrict__ wsl) {
  __shared__ float g[Ee * Dd];                       // 32 KB gate weights
  const int tid = threadIdx.x;
  for (int i = tid; i < Ee * Dd / 4; i += 256)
    ((float4*)g)[i] = ((const float4*)gw)[i];
  __syncthreads();

  const int lane = tid & 31, wid = tid >> 5;
  const int t = blockIdx.x * 8 + wid;
  const float4* xr = (const float4*)(x + (size_t)t * Dd);
  uint2*        xo = (uint2*)(xb + (size_t)t * Dd);

  float acc[Ee];
#pragma unroll
  for (int e = 0; e < Ee; ++e) acc[e] = 0.f;

#pragma unroll
  for (int i = 0; i < Dd / 128; ++i) {               // 8 iterations, float4 each
    const int idx = i * 32 + lane;
    float4 v = xr[idx];
    uint2 p;
    p.x = (uint32_t)f2bf(v.x) | ((uint32_t)f2bf(v.y) << 16);
    p.y = (uint32_t)f2bf(v.z) | ((uint32_t)f2bf(v.w) << 16);
    xo[idx] = p;
    const int base = idx * 4;
#pragma unroll
    for (int e = 0; e < Ee; ++e) {
      const float* ge = g + e * Dd + base;
      acc[e] += v.x * ge[0] + v.y * ge[1] + v.z * ge[2] + v.w * ge[3];
    }
  }
#pragma unroll
  for (int e = 0; e < Ee; ++e)
#pragma unroll
    for (int off = 16; off > 0; off >>= 1)
      acc[e] += __shfl_xor(acc[e], off, 32);

  if (lane == 0) {
    int i0 = 0; float s0 = acc[0];
#pragma unroll
    for (int e = 1; e < Ee; ++e) if (acc[e] > s0) { s0 = acc[e]; i0 = e; }
    int i1 = -1; float s1 = -3.4e38f;
#pragma unroll
    for (int e = 0; e < Ee; ++e) if (e != i0 && acc[e] > s1) { s1 = acc[e]; i1 = e; }
    const float z  = __expf(s1 - s0);
    const float w0 = __builtin_amdgcn_rcpf(1.f + z);
    const float w1 = z * w0;
    const int slot0 = t * 2, slot1 = t * 2 + 1;
    wsl[slot0] = w0; wsl[slot1] = w1;
    int p0 = atomicAdd(&cnt[i0], 1); list[i0 * Tn + p0] = slot0;
    int p1 = atomicAdd(&cnt[i1], 1); list[i1 * Tn + p1] = slot1;
  }
}

// ---------------------------------------------------------------------------
// WMMA fragment helpers (layouts per CDNA5 ISA §7.12.2).
union FragU { v16bf v; uint4 q[2]; };

#define BKP 40   // 32-halves K tile + 8-half pad (bank spread), rows 16B-aligned

// A frag: lane L (lo: M=L, K 0..7 & 16..23; hi: M=L-16, K 8..15 & 24..31)
__device__ __forceinline__ v16bf load_a_frag(const uint16_t* base, int llo, int lhi) {
  FragU f;
  const uint16_t* p = base + llo * BKP + lhi * 8;
  f.q[0] = *(const uint4*)p;
  f.q[1] = *(const uint4*)(p + 16);
  return f.v;
}
// B frag: lane L holds N=L&15 (lo: K 0..7 & 16..23; hi: K 8..15 & 24..31) (tile stored [N][K])
__device__ __forceinline__ v16bf load_b_frag(const uint16_t* base, int llo, int lhi) {
  FragU f;
  const uint16_t* p = base + llo * BKP + lhi * 8;
  f.q[0] = *(const uint4*)p;
  f.q[1] = *(const uint4*)(p + 16);
  return f.v;
}

// ---------------------------------------------------------------------------
// Up projection: h[slot, :] = silu(x@w1) * (x@w3) for tokens routed to expert e.
// Block tile 128x128xK32, 8 waves, wave tile 64x32, two GEMMs share the A frag.
// Async global->LDS double buffering (no VGPR staging -> no spills).
__global__ __launch_bounds__(256) void gemm_up_kernel(
    const uint16_t* __restrict__ xb, const uint16_t* __restrict__ w1t,
    const uint16_t* __restrict__ w3t, const int* __restrict__ cnt,
    const int* __restrict__ list, uint16_t* __restrict__ hbuf) {
  const int e  = blockIdx.z;
  const int m0 = blockIdx.x * 128;
  const int n0 = blockIdx.y * 128;            // over H
  const int M  = cnt[e];
  if (m0 >= M) return;
  const int* lst = list + e * Tn;

  __shared__ uint16_t sa [NBUF][128 * BKP];
  __shared__ uint16_t sb1[NBUF][128 * BKP];
  __shared__ uint16_t sb3[NBUF][128 * BKP];

  const int tid = threadIdx.x;
  const int lr = tid >> 1, lp = tid & 1;      // load row / 32B half-row
  const int ldst = lr * BKP + lp * 16;        // per-thread LDS dest (halves)
  const int grow = m0 + lr;
  const int aslot = lst[(grow < M) ? grow : 0];
  const uint16_t* arow  = xb  + (size_t)(aslot >> 1) * Dd + lp * 16;
  const uint16_t* b1row = w1t + ((size_t)e * Hh + (n0 + lr)) * Dd + lp * 16;
  const uint16_t* b3row = w3t + ((size_t)e * Hh + (n0 + lr)) * Dd + lp * 16;

  const int lane = tid & 31, wid = tid >> 5;
  const int wm = (wid & 1) * 64, wn = (wid >> 1) * 32;
  const int llo = lane & 15, lhi = lane >> 4;

  const v8f vz = {};
  v8f acc1[4][2], acc3[4][2];
#pragma unroll
  for (int mi = 0; mi < 4; ++mi)
#pragma unroll
    for (int ni = 0; ni < 2; ++ni) { acc1[mi][ni] = vz; acc3[mi][ni] = vz; }

#if HAVE_ASYNC
  // prologue: async-stage tile k=0 into buffer 0
  async_cp16(&sa [0][ldst], arow);
  async_cp16(&sb1[0][ldst], b1row);
  async_cp16(&sb3[0][ldst], b3row);
#endif
  int cur = 0;

  for (int k0 = 0; k0 < Dd; k0 += 32) {
#if HAVE_ASYNC
    async_wait0();                 // my async copies into buf[cur] are done
    __syncthreads();               // everyone's are; reads of buf[cur^1] done
    if (k0 + 32 < Dd) {            // stage next tile into the other buffer
      const int nxt = cur ^ 1;
      async_cp16(&sa [nxt][ldst], arow  + k0 + 32);
      async_cp16(&sb1[nxt][ldst], b1row + k0 + 32);
      async_cp16(&sb3[nxt][ldst], b3row + k0 + 32);
    }
#else
    __syncthreads();
    {
      const uint4* ga  = (const uint4*)(arow  + k0);
      const uint4* gb1 = (const uint4*)(b1row + k0);
      const uint4* gb3 = (const uint4*)(b3row + k0);
      uint4* da  = (uint4*)(&sa [0][ldst]);
      uint4* db1 = (uint4*)(&sb1[0][ldst]);
      uint4* db3 = (uint4*)(&sb3[0][ldst]);
      da [0] = ga [0]; da [1] = ga [1];
      db1[0] = gb1[0]; db1[1] = gb1[1];
      db3[0] = gb3[0]; db3[1] = gb3[1];
    }
    __syncthreads();
#endif

    const uint16_t* csa  = sa [cur];
    const uint16_t* csb1 = sb1[cur];
    const uint16_t* csb3 = sb3[cur];

    v16bf b1f[2], b3f[2];
#pragma unroll
    for (int ni = 0; ni < 2; ++ni) {
      b1f[ni] = load_b_frag(csb1 + (wn + ni * 16) * BKP, llo, lhi);
      b3f[ni] = load_b_frag(csb3 + (wn + ni * 16) * BKP, llo, lhi);
    }
#pragma unroll
    for (int mi = 0; mi < 4; ++mi) {
      v16bf af = load_a_frag(csa + (wm + mi * 16) * BKP, llo, lhi);
#pragma unroll
      for (int ni = 0; ni < 2; ++ni) {
        acc1[mi][ni] = __builtin_amdgcn_wmma_f32_16x16x32_bf16(
            false, af, false, b1f[ni], (short)0, acc1[mi][ni], false, false);
        acc3[mi][ni] = __builtin_amdgcn_wmma_f32_16x16x32_bf16(
            false, af, false, b3f[ni], (short)0, acc3[mi][ni], false, false);
      }
    }
    cur ^= (NBUF - 1);
  }

  // epilogue: h = silu(c1) * c3 = c1 * c3 * rcp(1 + e^-c1), store bf16 per slot
#pragma unroll
  for (int mi = 0; mi < 4; ++mi)
#pragma unroll
    for (int ni = 0; ni < 2; ++ni)
#pragma unroll
      for (int r = 0; r < 8; ++r) {
        const int row = m0 + wm + mi * 16 + r + lhi * 8;
        if (row < M) {
          const float c1  = acc1[mi][ni][r];
          const float c3  = acc3[mi][ni][r];
          const float sig = __builtin_amdgcn_rcpf(1.f + __expf(-c1));
          const float h   = c1 * sig * c3;
          const int slot  = lst[row];
          const int col   = n0 + wn + ni * 16 + llo;
          hbuf[(size_t)slot * Hh + col] = f2bf(h);
        }
      }
}

// ---------------------------------------------------------------------------
// Down projection: partial[slot, :] = routing_weight * (h[slot] @ w2[e]).
__global__ __launch_bounds__(256) void gemm_down_kernel(
    const uint16_t* __restrict__ hbuf, const uint16_t* __restrict__ w2t,
    const int* __restrict__ cnt, const int* __restrict__ list,
    const float* __restrict__ wsl, float* __restrict__ part) {
  const int e  = blockIdx.z;
  const int m0 = blockIdx.x * 128;
  const int n0 = blockIdx.y * 128;            // over D
  const int M  = cnt[e];
  if (m0 >= M) return;
  const int* lst = list + e * Tn;

  __shared__ uint16_t sa[NBUF][128 * BKP];
  __shared__ uint16_t sb[NBUF][128 * BKP];

  const int tid = threadIdx.x;
  const int lr = tid >> 1, lp = tid & 1;
  const int ldst = lr * BKP + lp * 16;
  const int grow = m0 + lr;
  const int aslot = lst[(grow < M) ? grow : 0];
  const uint16_t* arow = hbuf + (size_t)aslot * Hh + lp * 16;
  const uint16_t* brow = w2t + ((size_t)e * Dd + (n0 + lr)) * Hh + lp * 16;

  const int lane = tid & 31, wid = tid >> 5;
  const int wm = (wid & 1) * 64, wn = (wid >> 1) * 32;
  const int llo = lane & 15, lhi = lane >> 4;

  const v8f vz = {};
  v8f acc[4][2];
#pragma unroll
  for (int mi = 0; mi < 4; ++mi)
#pragma unroll
    for (int ni = 0; ni < 2; ++ni) acc[mi][ni] = vz;

#if HAVE_ASYNC
  async_cp16(&sa[0][ldst], arow);
  async_cp16(&sb[0][ldst], brow);
#endif
  int cur = 0;

  for (int k0 = 0; k0 < Hh; k0 += 32) {
#if HAVE_ASYNC
    async_wait0();
    __syncthreads();
    if (k0 + 32 < Hh) {
      const int nxt = cur ^ 1;
      async_cp16(&sa[nxt][ldst], arow + k0 + 32);
      async_cp16(&sb[nxt][ldst], brow + k0 + 32);
    }
#else
    __syncthreads();
    {
      const uint4* ga = (const uint4*)(arow + k0);
      const uint4* gb = (const uint4*)(brow + k0);
      uint4* da = (uint4*)(&sa[0][ldst]);
      uint4* db = (uint4*)(&sb[0][ldst]);
      da[0] = ga[0]; da[1] = ga[1];
      db[0] = gb[0]; db[1] = gb[1];
    }
    __syncthreads();
#endif

    const uint16_t* csa = sa[cur];
    const uint16_t* csb = sb[cur];

    v16bf bf[2];
#pragma unroll
    for (int ni = 0; ni < 2; ++ni)
      bf[ni] = load_b_frag(csb + (wn + ni * 16) * BKP, llo, lhi);
#pragma unroll
    for (int mi = 0; mi < 4; ++mi) {
      v16bf af = load_a_frag(csa + (wm + mi * 16) * BKP, llo, lhi);
#pragma unroll
      for (int ni = 0; ni < 2; ++ni)
        acc[mi][ni] = __builtin_amdgcn_wmma_f32_16x16x32_bf16(
            false, af, false, bf[ni], (short)0, acc[mi][ni], false, false);
    }
    cur ^= (NBUF - 1);
  }

#pragma unroll
  for (int mi = 0; mi < 4; ++mi)
#pragma unroll
    for (int ni = 0; ni < 2; ++ni)
#pragma unroll
      for (int r = 0; r < 8; ++r) {
        const int row = m0 + wm + mi * 16 + r + lhi * 8;
        if (row < M) {
          const int slot = lst[row];
          const int col  = n0 + wn + ni * 16 + llo;
          part[(size_t)slot * Dd + col] = acc[mi][ni][r] * wsl[slot];
        }
      }
}

// ---------------------------------------------------------------------------
// out[t] = partial[2t] + partial[2t+1]   (float4 vectorized)
__global__ void combine_kernel(const float* __restrict__ part,
                               float* __restrict__ out) {
  const size_t i   = (size_t)blockIdx.x * blockDim.x + threadIdx.x; // float4 idx
  const size_t col = i & (Dd / 4 - 1);
  const size_t t   = i >> 8;                 // i / (D/4)
  const float4* p  = (const float4*)part;
  float4 a = p[(t * 2)     * (Dd / 4) + col];
  float4 b = p[(t * 2 + 1) * (Dd / 4) + col];
  float4 o; o.x = a.x + b.x; o.y = a.y + b.y; o.z = a.z + b.z; o.w = a.w + b.w;
  ((float4*)out)[i] = o;
}

// ---------------------------------------------------------------------------
extern "C" void kernel_launch(void* const* d_in, const int* in_sizes, int n_in,
                              void* d_out, int out_size, void* d_ws, size_t ws_size,
                              hipStream_t stream) {
  const float* x  = (const float*)d_in[0];
  const float* gw = (const float*)d_in[1];
  const float* w1 = (const float*)d_in[2];
  const float* w2 = (const float*)d_in[3];
  const float* w3 = (const float*)d_in[4];

  char* ws = (char*)d_ws;
  uint16_t* xb   = (uint16_t*)(ws + OFF_XB);
  uint16_t* w1t  = (uint16_t*)(ws + OFF_W1T);
  uint16_t* w3t  = (uint16_t*)(ws + OFF_W3T);
  uint16_t* w2t  = (uint16_t*)(ws + OFF_W2T);
  uint16_t* hbuf = (uint16_t*)(ws + OFF_HBUF);
  float*    part = (float*)   (ws + OFF_PART);
  int*      cnt  = (int*)     (ws + OFF_CNT);
  int*      list = (int*)     (ws + OFF_LIST);
  float*    wsl  = (float*)   (ws + OFF_WSL);

  zero_counts_kernel<<<1, 64, 0, stream>>>(cnt);

  // weight convert+transpose (bf16 weights become L2-resident, ~138MB < 192MB)
  transpose_bf16_kernel<<<dim3(Hh / 32, Dd / 32, Ee), dim3(32, 8), 0, stream>>>(w1, w1t, Dd, Hh);
  transpose_bf16_kernel<<<dim3(Hh / 32, Dd / 32, Ee), dim3(32, 8), 0, stream>>>(w3, w3t, Dd, Hh);
  transpose_bf16_kernel<<<dim3(Dd / 32, Hh / 32, Ee), dim3(32, 8), 0, stream>>>(w2, w2t, Hh, Dd);

  router_kernel<<<Tn / 8, 256, 0, stream>>>(x, gw, xb, cnt, list, wsl);

  gemm_up_kernel  <<<dim3(Tn / 128, Hh / 128, Ee), 256, 0, stream>>>(xb, w1t, w3t, cnt, list, hbuf);
  gemm_down_kernel<<<dim3(Tn / 128, Dd / 128, Ee), 256, 0, stream>>>(hbuf, w2t, cnt, list, wsl, part);

  combine_kernel<<<(Tn * (size_t)Dd / 4) / 256, 256, 0, stream>>>(part, (float*)d_out);
}
